// SeemoRe_57561151701468
// MI455X (gfx1250) — matrix operand order repaired and checked by reading, not compiled, not run on hardware
//
#include <hip/hip_runtime.h>
#include <math.h>

// ---------------------------------------------------------------------------
// CDNA5 WMMA (wave32): f16 inputs, fp32 accumulate, 16x16x32 shape.
// f16 activations are stored PIXEL-MAJOR interleaved: addr = p*cp + cof + ci,
// so every WMMA B fragment is one contiguous 32B (v16h) load.
// ---------------------------------------------------------------------------
typedef __attribute__((ext_vector_type(16))) _Float16 v16h;
typedef __attribute__((ext_vector_type(8)))  _Float16 v8h;
typedef __attribute__((ext_vector_type(8)))  float    v8f;

__device__ __forceinline__ float gelu_f(float v) {
    return 0.5f * v * (1.0f + erff(v * 0.70710678118654752f));
}

// ---------------------------------------------------------------------------
// Weight pre-swizzle: fp32 conv weight [COUT][64] (x taps) -> f16 in exact
// WMMA A-fragment order, so an A fragment is one contiguous 32B (v16h) load.
// dest[tap*COUT*64 + (mrow*4 + half*2 + kc)*16 + e]
//   = W[(mrow*64 + ci)*taps + tap],  ci = kc*32 + half*16 + e
// ---------------------------------------------------------------------------
__global__ void prep_weights_kernel(const float* __restrict__ W,
                                    _Float16* __restrict__ A, int COUT, int taps)
{
    int idx = blockIdx.x * blockDim.x + threadIdx.x;
    int total = taps * COUT * 64;
    if (idx >= total) return;
    int tap  = idx / (COUT * 64);
    int rem  = idx - tap * COUT * 64;
    int mrow = rem >> 6;
    int t    = rem & 63;
    int half = t >> 5;
    int kc   = (t >> 4) & 1;
    int e    = t & 15;
    int ci   = kc * 32 + half * 16 + e;
    A[tap * COUT * 64 + (mrow * 4 + half * 2 + kc) * 16 + e] =
        (_Float16)W[(mrow * 64 + ci) * taps + tap];
}

// D-store helpers (overloaded on output type):
//  - fp32 trunk: channel-major [co][p] scalar stores (residual path)
//  - f16: pixel-major interleaved, one packed 16B store per (m,half)
__device__ __forceinline__ void gemm_store(float* __restrict__ Y, int P, int cpY, int cofY,
                                           int n, int m, int half, const float* tv)
{
#pragma unroll
    for (int r = 0; r < 8; ++r) Y[(size_t)(m * 16 + r + half * 8) * P + n] = tv[r];
}
__device__ __forceinline__ void gemm_store(_Float16* __restrict__ Y, int P, int cpY, int cofY,
                                           int n, int m, int half, const float* tv)
{
    v8h h;
#pragma unroll
    for (int r = 0; r < 8; ++r) h[r] = (_Float16)tv[r];
    *(v8h*)(Y + (size_t)n * cpY + cofY + m * 16 + half * 8) = h;
}

// ---------------------------------------------------------------------------
// WMMA 1x1-conv GEMM: Y[co,p] = act( sum_ci W[co,ci]*B(ci,p) + bias[co] ) (+Res)
// B(ci,p) = X[p*cpX+cofX+ci] * (MULB ? X2[p*cpX2+cofX2+ci] : 1)
// ---------------------------------------------------------------------------
template <int COUT, bool MULB, bool GELU, bool RES, typename TOUT>
__global__ __launch_bounds__(256) void gemm1x1_wmma(
    const _Float16* __restrict__ X, int cpX, int cofX,
    const _Float16* __restrict__ X2, int cpX2, int cofX2,
    const _Float16* __restrict__ A, const float* __restrict__ Bias,
    const float* __restrict__ Res, TOUT* __restrict__ Y, int cpY, int cofY, int P)
{
    const int lane = threadIdx.x & 31;
    const int half = lane >> 4;
    const int nl   = lane & 15;
    const int tile = blockIdx.x * (blockDim.x >> 5) + (threadIdx.x >> 5);
    const int p0   = tile * 16;
    if (p0 >= P) return;                 // wave-uniform; EXEC stays full
    const int n = p0 + nl;

    const _Float16* bx  = X + (size_t)n * cpX + cofX;
    const _Float16* bx2 = MULB ? (X2 + (size_t)n * cpX2 + cofX2) : (const _Float16*)0;

    v8f acc[COUT / 16];
    const v8f zero = {0.f, 0.f, 0.f, 0.f, 0.f, 0.f, 0.f, 0.f};
#pragma unroll
    for (int m = 0; m < COUT / 16; ++m) acc[m] = zero;

#pragma unroll
    for (int kc = 0; kc < 2; ++kc) {
        v16h bfrag = *(const v16h*)(bx + kc * 32 + half * 16);
        if (MULB) {
            v16h b2 = *(const v16h*)(bx2 + kc * 32 + half * 16);
            bfrag = bfrag * b2;          // v_pk_mul_f16 x8
        }
#pragma unroll
        for (int m = 0; m < COUT / 16; ++m) {
            const v16h afrag =
                *(const v16h*)(A + (((m * 16 + nl) * 4 + half * 2 + kc) << 4));
            acc[m] = __builtin_amdgcn_wmma_f32_16x16x32_f16(
                false, afrag, false, bfrag, (short)0, acc[m], false, false);
        }
    }
#pragma unroll
    for (int m = 0; m < COUT / 16; ++m) {
        float tv[8];
#pragma unroll
        for (int r = 0; r < 8; ++r) {
            const int co = m * 16 + r + half * 8;
            float v = acc[m][r] + Bias[co];
            if (GELU) v = gelu_f(v);
            if (RES)  v += Res[(size_t)co * P + n];
            tv[r] = v;
        }
        gemm_store(Y, P, cpY, cofY, n, m, half, tv);
    }
}

// ---------------------------------------------------------------------------
// Dense 3x3 conv 64->64, pad 1, fused GELU: 9 shifted WMMA GEMMs.
// Input is ZERO-PADDED (H+2)x(W+2), interleaved [py*(W+2)+px][64] f16, so all
// tap loads are unconditional contiguous v16h loads. Output [p][64] f16.
// ---------------------------------------------------------------------------
__global__ __launch_bounds__(256) void conv3x3_wmma_gelu(
    const _Float16* __restrict__ Xp, const _Float16* __restrict__ A9,
    const float* __restrict__ Bias, _Float16* __restrict__ Y, int H, int Wd)
{
    const int P    = H * Wd;
    const int Wp   = Wd + 2;
    const int lane = threadIdx.x & 31;
    const int half = lane >> 4;
    const int nl   = lane & 15;
    const int tile = blockIdx.x * (blockDim.x >> 5) + (threadIdx.x >> 5);
    const int p0   = tile * 16;
    if (p0 >= P) return;
    const int oy  = p0 / Wd;             // Wd % 16 == 0 -> tile stays on one row
    const int ox0 = p0 - oy * Wd;
    const int n   = p0 + nl;

    v8f acc[4];
    const v8f zero = {0.f, 0.f, 0.f, 0.f, 0.f, 0.f, 0.f, 0.f};
#pragma unroll
    for (int m = 0; m < 4; ++m) acc[m] = zero;

#pragma unroll
    for (int ty = 0; ty < 3; ++ty) {
#pragma unroll
        for (int tx = 0; tx < 3; ++tx) {
            // source pixel (oy+ty-1, ox+tx-1) -> padded (+1,+1) = (oy+ty, ox+tx)
            const _Float16* bx =
                Xp + (size_t)((oy + ty) * Wp + ox0 + nl + tx) * 64;
            const _Float16* Atap = A9 + (ty * 3 + tx) * 4096;
#pragma unroll
            for (int kc = 0; kc < 2; ++kc) {
                const v16h bfrag = *(const v16h*)(bx + kc * 32 + half * 16);
#pragma unroll
                for (int m = 0; m < 4; ++m) {
                    const v16h afrag =
                        *(const v16h*)(Atap + (((m * 16 + nl) * 4 + half * 2 + kc) << 4));
                    acc[m] = __builtin_amdgcn_wmma_f32_16x16x32_f16(
                        false, afrag, false, bfrag, (short)0, acc[m], false, false);
                }
            }
        }
    }
#pragma unroll
    for (int m = 0; m < 4; ++m) {
        v8h h;
#pragma unroll
        for (int r = 0; r < 8; ++r)
            h[r] = (_Float16)gelu_f(acc[m][r] + Bias[m * 16 + r + half * 8]);
        *(v8h*)(Y + (size_t)n * 64 + m * 16 + half * 8) = h;
    }
}

// ---------------------------------------------------------------------------
// Zero the halo of the padded interleaved buffer (once per launch).
// ---------------------------------------------------------------------------
__global__ void zero_pad_kernel(_Float16* __restrict__ Xp, int H, int Wd)
{
    const int Wp = Wd + 2, Hp = H + 2;
    const int npix = 2 * Wp + 2 * H;
    int idx = blockIdx.x * blockDim.x + threadIdx.x;
    if (idx >= npix * 64) return;
    int pix = idx >> 6, c = idx & 63;
    int py, px;
    if (pix < Wp)            { py = 0;      px = pix; }
    else if (pix < 2 * Wp)   { py = Hp - 1; px = pix - Wp; }
    else { int r2 = pix - 2 * Wp; py = 1 + (r2 >> 1); px = (r2 & 1) ? (Wp - 1) : 0; }
    Xp[(size_t)(py * Wp + px) * 64 + c] = (_Float16)0.0f;
}

// ---------------------------------------------------------------------------
// LayerNorm variants (input fp32 channel-major trunk)
// ---------------------------------------------------------------------------
__global__ void layernorm_il_kernel(const float* __restrict__ X, const float* __restrict__ g,
                                    const float* __restrict__ b, _Float16* __restrict__ Y, int P)
{
    int p = blockIdx.x * blockDim.x + threadIdx.x;
    if (p >= P) return;
    float s = 0.f, s2 = 0.f;
    for (int c = 0; c < 64; ++c) { float v = X[c * P + p]; s += v; s2 += v * v; }
    const float u   = s * (1.0f / 64.0f);
    const float var = s2 * (1.0f / 64.0f) - u * u;
    const float inv = rsqrtf(var + 1e-6f);
    _Float16* yp = Y + (size_t)p * 64;
    for (int c = 0; c < 64; ++c)
        yp[c] = (_Float16)(g[c] * (X[c * P + p] - u) * inv + b[c]);
}

__global__ void layernorm_pad_kernel(const float* __restrict__ X, const float* __restrict__ g,
                                     const float* __restrict__ b, _Float16* __restrict__ Yp,
                                     int Wd, int P)
{
    int p = blockIdx.x * blockDim.x + threadIdx.x;
    if (p >= P) return;
    int oy = p / Wd, ox = p - oy * Wd;
    float s = 0.f, s2 = 0.f;
    for (int c = 0; c < 64; ++c) { float v = X[c * P + p]; s += v; s2 += v * v; }
    const float u   = s * (1.0f / 64.0f);
    const float var = s2 * (1.0f / 64.0f) - u * u;
    const float inv = rsqrtf(var + 1e-6f);
    _Float16* yp = Yp + (size_t)((oy + 1) * (Wd + 2) + ox + 1) * 64;
    for (int c = 0; c < 64; ++c)
        yp[c] = (_Float16)(g[c] * (X[c * P + p] - u) * inv + b[c]);
}

__global__ void layernorm_f32_kernel(const float* __restrict__ X, const float* __restrict__ g,
                                     const float* __restrict__ b, float* __restrict__ Y, int P)
{
    int p = blockIdx.x * blockDim.x + threadIdx.x;
    if (p >= P) return;
    float s = 0.f, s2 = 0.f;
    for (int c = 0; c < 64; ++c) { float v = X[c * P + p]; s += v; s2 += v * v; }
    const float u   = s * (1.0f / 64.0f);
    const float var = s2 * (1.0f / 64.0f) - u * u;
    const float inv = rsqrtf(var + 1e-6f);
    for (int c = 0; c < 64; ++c)
        Y[c * P + p] = g[c] * (X[c * P + p] - u) * inv + b[c];
}

// ---------------------------------------------------------------------------
// Depthwise conv; IL flags select pixel-major-interleaved (cp/cof) addressing
// ---------------------------------------------------------------------------
template <typename TI, typename TO, bool ILI, bool ILO>
__global__ void dwconv_kernel(const TI* __restrict__ X, const float* __restrict__ Wt,
                              const float* __restrict__ B, TO* __restrict__ Y,
                              int C, int Hi, int Wi, int Ho, int Wo,
                              int kh, int kw, int st, int ph, int pw2, int do_gelu,
                              int cpI, int cofI, int cpO, int cofO)
{
    int idx = blockIdx.x * blockDim.x + threadIdx.x;
    int total = C * Ho * Wo;
    if (idx >= total) return;
    int c = idx / (Ho * Wo);
    int r = idx - c * (Ho * Wo);
    int oy = r / Wo, ox = r - (r / Wo) * Wo;
    float acc = B[c];
    for (int ky = 0; ky < kh; ++ky) {
        int sy = oy * st - ph + ky;
        if (sy < 0 || sy >= Hi) continue;
        for (int kx = 0; kx < kw; ++kx) {
            int sx = ox * st - pw2 + kx;
            if (sx < 0 || sx >= Wi) continue;
            float xv = ILI ? (float)X[(size_t)(sy * Wi + sx) * cpI + cofI + c]
                           : (float)X[(size_t)(c * Hi + sy) * Wi + sx];
            acc += Wt[(c * kh + ky) * kw + kx] * xv;
        }
    }
    if (do_gelu) acc = gelu_f(acc);
    if (ILO) Y[(size_t)(oy * Wo + ox) * cpO + cofO + c] = (TO)acc;
    else     Y[idx] = (TO)acc;
}

__global__ void conv1x1_small_kernel(const float* __restrict__ X, const float* __restrict__ Wt,
                                     const float* __restrict__ B, float* __restrict__ Y,
                                     int Cin, int Cout, int P)
{
    int idx = blockIdx.x * blockDim.x + threadIdx.x;
    if (idx >= Cout * P) return;
    int co = idx / P, p = idx - co * P;
    float acc = B[co];
    for (int ci = 0; ci < Cin; ++ci) acc += Wt[co * Cin + ci] * X[ci * P + p];
    Y[idx] = acc;
}

// out(c-major fp32) = Kb(f16 interleaved) + bilinear_upsample(S fp32 c-major)
__global__ void bilinear_add_kernel(const _Float16* __restrict__ Kb, int cpK, int cofK,
                                    const float* __restrict__ S,
                                    float* __restrict__ Y, int C, int Hs, int Ws, int Hb, int Wb)
{
    int idx = blockIdx.x * blockDim.x + threadIdx.x;
    if (idx >= C * Hb * Wb) return;
    int c = idx / (Hb * Wb);
    int r = idx - c * (Hb * Wb);
    int oy = r / Wb, ox = r - (r / Wb) * Wb;
    float sy = (oy + 0.5f) * ((float)Hs / (float)Hb) - 0.5f;
    float sx = (ox + 0.5f) * ((float)Ws / (float)Wb) - 0.5f;
    sy = fminf(fmaxf(sy, 0.f), (float)(Hs - 1));
    sx = fminf(fmaxf(sx, 0.f), (float)(Ws - 1));
    int y0 = (int)sy, x0 = (int)sx;
    int y1 = (y0 < Hs - 1) ? y0 + 1 : Hs - 1;
    int x1 = (x0 < Ws - 1) ? x0 + 1 : Ws - 1;
    float fy = sy - (float)y0, fx = sx - (float)x0;
    const float* sp = S + c * Hs * Ws;
    float v = (1.f - fy) * ((1.f - fx) * sp[y0 * Ws + x0] + fx * sp[y0 * Ws + x1]) +
              fy * ((1.f - fx) * sp[y1 * Ws + x0] + fx * sp[y1 * Ws + x1]);
    Y[idx] = (float)Kb[(size_t)(oy * Wb + ox) * cpK + cofK + c] + v;
}

__global__ void pool_reduce_kernel(const float* __restrict__ X, float* __restrict__ pooled, int P)
{
    __shared__ float sm[256];
    int c = blockIdx.x;
    float s = 0.f;
    for (int p = threadIdx.x; p < P; p += blockDim.x) s += X[c * P + p];
    sm[threadIdx.x] = s;
    __syncthreads();
    for (int st = 128; st > 0; st >>= 1) {
        if ((int)threadIdx.x < st) sm[threadIdx.x] += sm[threadIdx.x + st];
        __syncthreads();
    }
    if (threadIdx.x == 0) pooled[c] = sm[0] / (float)P;
}

// softmax over 6 router logits, top-2 mask (first-index tiebreak like lax.top_k)
__global__ void router_kernel(const float* __restrict__ pooled, const float* __restrict__ R,
                              float* __restrict__ mask)
{
    if (threadIdx.x != 0 || blockIdx.x != 0) return;
    float lg[6];
    for (int e = 0; e < 6; ++e) {
        float s = 0.f;
        for (int c = 0; c < 64; ++c) s += pooled[c] * R[e * 64 + c];
        lg[e] = s;
    }
    float mx = lg[0];
    for (int e = 1; e < 6; ++e) mx = fmaxf(mx, lg[e]);
    float w[6], sum = 0.f;
    for (int e = 0; e < 6; ++e) { w[e] = expf(lg[e] - mx); sum += w[e]; }
    for (int e = 0; e < 6; ++e) w[e] /= sum;
    int i1 = 0;
    for (int e = 1; e < 6; ++e) if (w[e] > w[i1]) i1 = e;
    int i2 = -1;
    for (int e = 0; e < 6; ++e) { if (e == i1) continue; if (i2 < 0 || w[e] > w[i2]) i2 = e; }
    for (int e = 0; e < 6; ++e) mask[e] = (e == i1) ? w[i1] : ((e == i2) ? w[i2] : 0.f);
}

struct ExpertPtrs { const float *w1, *b1, *w2, *b2, *w3, *b3; };
struct ExpertPack { ExpertPtrs e[6]; };

// out (f16 interleaved [p][64]) = x + sum_e mask[e]*expert_e(x,k)
__global__ void moe_kernel(const float* __restrict__ X, const float* __restrict__ Kc,
                           const float* __restrict__ mask, ExpertPack ep,
                           _Float16* __restrict__ Y, int P)
{
    int p = blockIdx.x * blockDim.x + threadIdx.x;
    if (p >= P) return;
    float t1[27], t2[27];   // sum of expert dims 2..7 = 27, fully unrolled -> regs
    {
        int off = 0;
#pragma unroll
        for (int e = 0; e < 6; ++e) {
            const int d = e + 2;
#pragma unroll
            for (int j = 0; j < d; ++j) { t1[off + j] = ep.e[e].b1[j]; t2[off + j] = ep.e[e].b2[j]; }
            off += d;
        }
    }
    for (int c = 0; c < 64; ++c) {
        const float xc = X[c * P + p], kc = Kc[c * P + p];
        int off = 0;
#pragma unroll
        for (int e = 0; e < 6; ++e) {
            const int d = e + 2;
            const float* w1 = ep.e[e].w1;
            const float* w2 = ep.e[e].w2;
#pragma unroll
            for (int j = 0; j < d; ++j) {
                t1[off + j] += w1[j * 64 + c] * xc;
                t2[off + j] += w2[j * 64 + c] * kc;
            }
            off += d;
        }
    }
    float pr[27];
#pragma unroll
    for (int i = 0; i < 27; ++i) pr[i] = t1[i] * t2[i];
    float mk[6];
#pragma unroll
    for (int e = 0; e < 6; ++e) mk[e] = mask[e];
    _Float16* yp = Y + (size_t)p * 64;
    for (int c = 0; c < 64; ++c) {
        float acc = X[c * P + p];
        int off = 0;
#pragma unroll
        for (int e = 0; e < 6; ++e) {
            const int d = e + 2;
            float ea = ep.e[e].b3[c];
#pragma unroll
            for (int j = 0; j < d; ++j) ea += ep.e[e].w3[c * d + j] * pr[off + j];
            acc += mk[e] * ea;
            off += d;
        }
        yp[c] = (_Float16)acc;
    }
}

__global__ void head_conv_kernel(const float* __restrict__ X, const float* __restrict__ Wt,
                                 const float* __restrict__ B, float* __restrict__ Y, int H, int Wd)
{
    int P = H * Wd;
    int idx = blockIdx.x * blockDim.x + threadIdx.x;
    if (idx >= 64 * P) return;
    int co = idx / P, p = idx - co * P;
    int oy = p / Wd, ox = p - (p / Wd) * Wd;
    float acc = B[co];
    for (int ci = 0; ci < 3; ++ci)
        for (int ky = 0; ky < 3; ++ky) {
            int sy = oy + ky - 1;
            if (sy < 0 || sy >= H) continue;
            for (int kx = 0; kx < 3; ++kx) {
                int sx = ox + kx - 1;
                if (sx < 0 || sx >= Wd) continue;
                acc += Wt[((co * 3 + ci) * 3 + ky) * 3 + kx] * X[ci * P + sy * Wd + sx];
            }
        }
    Y[idx] = acc;
}

// up-conv 64->12 (3x3 pad1) on (A + Bres), output pixel-shuffled (r=2) to [3][2H][2W]
__global__ void upconv_shuffle_kernel(const float* __restrict__ A, const float* __restrict__ Bres,
                                      const float* __restrict__ Wt, const float* __restrict__ Bb,
                                      float* __restrict__ Out, int H, int Wd)
{
    int P = H * Wd;
    int idx = blockIdx.x * blockDim.x + threadIdx.x;
    if (idx >= 12 * P) return;
    int co = idx / P, p = idx - co * P;
    int oy = p / Wd, ox = p - (p / Wd) * Wd;
    float acc = Bb[co];
    for (int ci = 0; ci < 64; ++ci)
        for (int ky = 0; ky < 3; ++ky) {
            int sy = oy + ky - 1;
            if (sy < 0 || sy >= H) continue;
            for (int kx = 0; kx < 3; ++kx) {
                int sx = ox + kx - 1;
                if (sx < 0 || sx >= Wd) continue;
                int sp = sy * Wd + sx;
                acc += Wt[((co * 64 + ci) * 3 + ky) * 3 + kx] * (A[ci * P + sp] + Bres[ci * P + sp]);
            }
        }
    int c0 = co >> 2, r1 = (co >> 1) & 1, r2 = co & 1;
    Out[(c0 * 2 * H + oy * 2 + r1) * (2 * Wd) + ox * 2 + r2] = acc;
}

// ---------------------------------------------------------------------------
// Host side: parameter pytree walk (setup_inputs insertion order) + launches
// ---------------------------------------------------------------------------
struct CP { const float* w; const float* b; };
struct FFNP { CP fn1, gate, fn2; };
struct ExpertW { CP c1, c2, c3; };
struct MoeP { CP c1a, c1b, agg, dw, pw, s1, s2; const float* router; ExpertW ex[6]; CP proj; };
struct RmeP { CP n1; MoeP moe; CP n2; FFNP ffn; };
struct SmeP { CP n1, qv, a1, a2, proj, n2; FFNP ffn; };
struct GroupP { RmeP rme; SmeP sme; };
struct GW { _Float16 *c1a, *c1b, *proj, *fn1, *fn2, *qv, *sproj, *sfn1, *sfn2; };

static inline CP takeCP(void* const* d, int& i) {
    CP p; p.w = (const float*)d[i]; p.b = (const float*)d[i + 1]; i += 2; return p;
}
static inline FFNP takeFFN(void* const* d, int& i) {
    FFNP f; f.fn1 = takeCP(d, i); f.gate = takeCP(d, i); f.fn2 = takeCP(d, i); return f;
}

extern "C" void kernel_launch(void* const* d_in, const int* in_sizes, int n_in,
                              void* d_out, int out_size, void* d_ws, size_t ws_size,
                              hipStream_t stream)
{
    (void)in_sizes; (void)n_in; (void)out_size; (void)ws_size;
    const int H = 160, Wd = 160, P = H * Wd;

    int i = 0;
    const float* x_in = (const float*)d_in[i++];
    CP head = takeCP(d_in, i);
    GroupP gp[4];
    for (int g = 0; g < 4; ++g) {
        RmeP& r = gp[g].rme;
        r.n1 = takeCP(d_in, i);
        MoeP& m = r.moe;
        m.c1a = takeCP(d_in, i); m.c1b = takeCP(d_in, i); m.agg = takeCP(d_in, i);
        m.dw  = takeCP(d_in, i); m.pw  = takeCP(d_in, i);
        m.s1  = takeCP(d_in, i); m.s2  = takeCP(d_in, i);
        m.router = (const float*)d_in[i++];
        for (int e = 0; e < 6; ++e) {
            m.ex[e].c1 = takeCP(d_in, i); m.ex[e].c2 = takeCP(d_in, i); m.ex[e].c3 = takeCP(d_in, i);
        }
        m.proj = takeCP(d_in, i);
        r.n2 = takeCP(d_in, i);
        r.ffn = takeFFN(d_in, i);
        SmeP& s = gp[g].sme;
        s.n1 = takeCP(d_in, i); s.qv = takeCP(d_in, i);
        s.a1 = takeCP(d_in, i); s.a2 = takeCP(d_in, i);
        s.proj = takeCP(d_in, i); s.n2 = takeCP(d_in, i);
        s.ffn = takeFFN(d_in, i);
    }
    CP norm = takeCP(d_in, i);
    CP up   = takeCP(d_in, i);

    // ---------------- workspace carve (32B-aligned blocks) ----------------
    char* base = (char*)d_ws;
    size_t off = 0;
    auto carve = [&](size_t bytes) -> void* {
        void* p = (void*)(base + off);
        off += (bytes + 31) & ~(size_t)31;
        return p;
    };
    const size_t NP = (size_t)64 * P;
    float* bufX = (float*)carve(NP * 4);          // running trunk (fp32, [c][p])
    float* bufR = (float*)carve(NP * 4);          // long residual
    float* T0   = (float*)carve(NP * 4);          // final LN out (fp32)
    float* T3   = (float*)carve(NP * 4);          // moe "x" (fp32)
    float* T4   = (float*)carve(NP * 4);          // calibrated k (fp32)
    float* S40  = (float*)carve((size_t)64 * 1600 * 4);
    float* S10a = (float*)carve(6400 * 4);
    float* S10b = (float*)carve(6400 * 4);
    float* S10c = (float*)carve(6400 * 4);
    float* pooled = (float*)carve(64 * 4);
    float* maskp  = (float*)carve(8 * 4);
    // f16 activation buffers, pixel-major interleaved
    _Float16* Hpad = (_Float16*)carve((size_t)(H + 2) * (Wd + 2) * 64 * 2);
    _Float16* H0   = (_Float16*)carve(NP * 2);
    _Float16* H1   = (_Float16*)carve(NP * 2);
    _Float16* H2   = (_Float16*)carve(NP * 2);
    _Float16* H128 = (_Float16*)carve((size_t)128 * P * 2);
    GW gw[4];
    for (int g = 0; g < 4; ++g) {
        gw[g].c1a   = (_Float16*)carve((size_t)9 * 64 * 64 * 2);
        gw[g].c1b   = (_Float16*)carve((size_t)128 * 64 * 2);
        gw[g].proj  = (_Float16*)carve((size_t)64 * 64 * 2);
        gw[g].fn1   = (_Float16*)carve((size_t)128 * 64 * 2);
        gw[g].fn2   = (_Float16*)carve((size_t)64 * 64 * 2);
        gw[g].qv    = (_Float16*)carve((size_t)128 * 64 * 2);
        gw[g].sproj = (_Float16*)carve((size_t)64 * 64 * 2);
        gw[g].sfn1  = (_Float16*)carve((size_t)128 * 64 * 2);
        gw[g].sfn2  = (_Float16*)carve((size_t)64 * 64 * 2);
    }

    const int TPB = 256;
    const int gP  = (P + TPB - 1) / TPB;
    const int gCP = (64 * P + TPB - 1) / TPB;
    const int gWM = P / (16 * 8);                 // 1600 N-tiles / 8 waves per block

    // ---------------- weight pre-swizzle (tiny; once per launch) ----------
    auto prep = [&](const float* W, _Float16* A, int COUT, int taps) {
        int total = taps * COUT * 64;
        prep_weights_kernel<<<(total + TPB - 1) / TPB, TPB, 0, stream>>>(W, A, COUT, taps);
    };
    for (int g = 0; g < 4; ++g) {
        const RmeP& r = gp[g].rme; const SmeP& s = gp[g].sme; const MoeP& m = r.moe;
        prep(m.c1a.w, gw[g].c1a, 64, 9);
        prep(m.c1b.w, gw[g].c1b, 128, 1);
        prep(m.proj.w, gw[g].proj, 64, 1);
        prep(r.ffn.fn1.w, gw[g].fn1, 128, 1);
        prep(r.ffn.fn2.w, gw[g].fn2, 64, 1);
        prep(s.qv.w, gw[g].qv, 128, 1);
        prep(s.proj.w, gw[g].sproj, 64, 1);
        prep(s.ffn.fn1.w, gw[g].sfn1, 128, 1);
        prep(s.ffn.fn2.w, gw[g].sfn2, 64, 1);
    }

    // zero the padded buffer's halo once; interiors are overwritten each group
    {
        int n = (2 * (Wd + 2) + 2 * H) * 64;
        zero_pad_kernel<<<(n + TPB - 1) / TPB, TPB, 0, stream>>>(Hpad, H, Wd);
    }

    head_conv_kernel<<<gCP, TPB, 0, stream>>>(x_in, head.w, head.b, bufX, H, Wd);
    hipMemcpyAsync(bufR, bufX, NP * sizeof(float), hipMemcpyDeviceToDevice, stream);

    for (int g = 0; g < 4; ++g) {
        const RmeP& r = gp[g].rme; const SmeP& s = gp[g].sme; const MoeP& m = r.moe;

        // ---------------- RME: moe_block(LN(x)) + x ----------------
        layernorm_pad_kernel<<<gP, TPB, 0, stream>>>(bufX, r.n1.w, r.n1.b, Hpad, Wd, P);
        conv3x3_wmma_gelu<<<gWM, TPB, 0, stream>>>(Hpad, gw[g].c1a, m.c1a.b, H1, H, Wd);
        gemm1x1_wmma<128, false, false, false, _Float16><<<gWM, TPB, 0, stream>>>(
            H1, 64, 0, nullptr, 0, 0, gw[g].c1b, m.c1b.b, nullptr, H128, 128, 0, P);
        // xs = H128[:,0:64], kk = H128[:,64:128] (interleaved views)
        dwconv_kernel<_Float16, _Float16, true, true><<<gCP, TPB, 0, stream>>>(
            H128, m.s1.w, m.s1.b, H0, 64, H, Wd, H, Wd, 1, 3, 1, 0, 1, 0, 128, 0, 64, 0);
        dwconv_kernel<_Float16, float, true, false><<<gCP, TPB, 0, stream>>>(
            H0, m.s2.w, m.s2.b, T3, 64, H, Wd, H, Wd, 3, 1, 1, 1, 0, 1, 64, 0, 0, 0);
        // calibrate(kk) -> T4
        dwconv_kernel<_Float16, float, true, false><<<(64 * 1600 + TPB - 1) / TPB, TPB, 0, stream>>>(
            H128, m.agg.w, m.agg.b, S40, 64, 160, 160, 40, 40, 4, 4, 4, 0, 0, 1, 128, 64, 0, 0);
        dwconv_kernel<float, float, false, false><<<(64 * 100 + TPB - 1) / TPB, TPB, 0, stream>>>(
            S40, m.agg.w, m.agg.b, S10a, 64, 40, 40, 10, 10, 4, 4, 4, 0, 0, 1, 0, 0, 0, 0);
        dwconv_kernel<float, float, false, false><<<(64 * 100 + TPB - 1) / TPB, TPB, 0, stream>>>(
            S10a, m.dw.w, m.dw.b, S10b, 64, 10, 10, 10, 10, 3, 3, 1, 1, 1, 0, 0, 0, 0, 0);
        conv1x1_small_kernel<<<(64 * 100 + TPB - 1) / TPB, TPB, 0, stream>>>(
            S10b, m.pw.w, m.pw.b, S10c, 64, 64, 100);
        bilinear_add_kernel<<<gCP, TPB, 0, stream>>>(H128, 128, 64, S10c, T4, 64, 10, 10, H, Wd);
        // moe_layer(T3, T4)
        pool_reduce_kernel<<<64, TPB, 0, stream>>>(T3, pooled, P);
        router_kernel<<<1, 32, 0, stream>>>(pooled, m.router, maskp);
        ExpertPack pack;
        for (int e = 0; e < 6; ++e) {
            pack.e[e].w1 = m.ex[e].c1.w; pack.e[e].b1 = m.ex[e].c1.b;
            pack.e[e].w2 = m.ex[e].c2.w; pack.e[e].b2 = m.ex[e].c2.b;
            pack.e[e].w3 = m.ex[e].c3.w; pack.e[e].b3 = m.ex[e].c3.b;
        }
        moe_kernel<<<gP, TPB, 0, stream>>>(T3, T4, maskp, pack, H0, P);
        gemm1x1_wmma<64, false, false, true, float><<<gWM, TPB, 0, stream>>>(
            H0, 64, 0, nullptr, 0, 0, gw[g].proj, m.proj.b, bufX, bufX, 0, 0, P);
        // gated_ffn(LN(x)) + x
        layernorm_il_kernel<<<gP, TPB, 0, stream>>>(bufX, r.n2.w, r.n2.b, H0, P);
        gemm1x1_wmma<128, false, true, false, _Float16><<<gWM, TPB, 0, stream>>>(
            H0, 64, 0, nullptr, 0, 0, gw[g].fn1, r.ffn.fn1.b, nullptr, H128, 128, 0, P);
        dwconv_kernel<_Float16, _Float16, true, true><<<gCP, TPB, 0, stream>>>(
            H128, r.ffn.gate.w, r.ffn.gate.b, H1, 64, H, Wd, H, Wd, 3, 3, 1, 1, 1, 0, 128, 64, 64, 0);
        gemm1x1_wmma<64, true, true, true, float><<<gWM, TPB, 0, stream>>>(
            H128, 128, 0, H1, 64, 0, gw[g].fn2, r.ffn.fn2.b, bufX, bufX, 0, 0, P);

        // ---------------- SME ----------------
        layernorm_il_kernel<<<gP, TPB, 0, stream>>>(bufX, s.n1.w, s.n1.b, H0, P);
        gemm1x1_wmma<128, false, true, false, _Float16><<<gWM, TPB, 0, stream>>>(
            H0, 64, 0, nullptr, 0, 0, gw[g].qv, s.qv.b, nullptr, H128, 128, 0, P);
        dwconv_kernel<_Float16, _Float16, true, true><<<gCP, TPB, 0, stream>>>(
            H128, s.a1.w, s.a1.b, H1, 64, H, Wd, H, Wd, 1, 11, 1, 0, 5, 0, 128, 0, 64, 0);
        dwconv_kernel<_Float16, _Float16, true, true><<<gCP, TPB, 0, stream>>>(
            H1, s.a2.w, s.a2.b, H2, 64, H, Wd, H, Wd, 11, 1, 1, 5, 0, 0, 64, 0, 64, 0);
        gemm1x1_wmma<64, true, false, true, float><<<gWM, TPB, 0, stream>>>(
            H2, 64, 0, H128, 128, 64, gw[g].sproj, s.proj.b, bufX, bufX, 0, 0, P);
        layernorm_il_kernel<<<gP, TPB, 0, stream>>>(bufX, s.n2.w, s.n2.b, H0, P);
        gemm1x1_wmma<128, false, true, false, _Float16><<<gWM, TPB, 0, stream>>>(
            H0, 64, 0, nullptr, 0, 0, gw[g].sfn1, s.ffn.fn1.b, nullptr, H128, 128, 0, P);
        dwconv_kernel<_Float16, _Float16, true, true><<<gCP, TPB, 0, stream>>>(
            H128, s.ffn.gate.w, s.ffn.gate.b, H1, 64, H, Wd, H, Wd, 3, 3, 1, 1, 1, 0, 128, 64, 64, 0);
        gemm1x1_wmma<64, true, true, true, float><<<gWM, TPB, 0, stream>>>(
            H128, 128, 0, H1, 64, 0, gw[g].sfn2, s.ffn.fn2.b, bufX, bufX, 0, 0, P);
    }

    layernorm_f32_kernel<<<gP, TPB, 0, stream>>>(bufX, norm.w, norm.b, T0, P);
    upconv_shuffle_kernel<<<(12 * P + TPB - 1) / TPB, TPB, 0, stream>>>(
        T0, bufR, up.w, up.b, (float*)d_out, H, Wd);
}
